// AFM_43722767073608
// MI455X (gfx1250) — matrix operand order repaired and checked by reading, not compiled, not run on hardware
//
#include <hip/hip_runtime.h>
#include <hip/hip_bf16.h>
#include <stdint.h>

typedef __attribute__((ext_vector_type(16))) _Float16 v16h;
typedef __attribute__((ext_vector_type(8)))  float    v8f;
typedef __attribute__((ext_vector_type(4)))  float    f4;

#define N_DENSE   13
#define N_SPARSE  26
#define VOCAB     100000
#define EMB_DIM   64
#define N_PAIR    325
#define N_TILE    21        // ceil(325/16)
#define TEMP_INV  10.0f

__device__ __forceinline__ void async_copy16(uint32_t ldsOff, const void* gsrc) {
    unsigned long long ga = (unsigned long long)(uintptr_t)gsrc;
    asm volatile("global_load_async_to_lds_b128 %0, %1, off"
                 :: "v"(ldsOff), "v"(ga) : "memory");
}

// sum across the 16 lanes of each half-wave group via ds_swizzle XOR mode
// (offset[15]=0: and_mask=0x1f, or_mask=0, xor_mask=m). masks 1/2/4/8 stay
// inside each 16-lane group (bit 4 untouched). EXEC is all-ones here.
__device__ __forceinline__ float group16_sum(float v) {
    v += __int_as_float(__builtin_amdgcn_ds_swizzle(__float_as_int(v), (1 << 10) | 0x1f));
    v += __int_as_float(__builtin_amdgcn_ds_swizzle(__float_as_int(v), (2 << 10) | 0x1f));
    v += __int_as_float(__builtin_amdgcn_ds_swizzle(__float_as_int(v), (4 << 10) | 0x1f));
    v += __int_as_float(__builtin_amdgcn_ds_swizzle(__float_as_int(v), (8 << 10) | 0x1f));
    return v;
}

__global__ __launch_bounds__(256)
void afm_fused_kernel(const float* __restrict__ dense_x,
                      const int*   __restrict__ sparse_idx,
                      const float* __restrict__ emb_tables,
                      const float* __restrict__ attn_W,
                      const float* __restrict__ attn_b,
                      const float* __restrict__ proj_W,
                      const float* __restrict__ proj_b,
                      const float* __restrict__ lin_W,
                      const float* __restrict__ lin_b,
                      const float* __restrict__ pred_W,
                      const float* __restrict__ pred_b,
                      float* __restrict__ out)
{
    // 27 rows: row 26 is an all-zero pad row for invalid pairs
    __shared__ __align__(16) float embL[(N_SPARSE + 1) * EMB_DIM];
    __shared__ __align__(16) float WsL[EMB_DIM * 12];   // attn_W 64x12
    __shared__ float logitsL[336];                      // padded to 21 tiles
    __shared__ float expL[N_PAIR];
    __shared__ float red[256];
    __shared__ float attRed[4 * 64];
    __shared__ float s2[64];
    __shared__ float biasL[16];
    __shared__ float projL[16];
    __shared__ unsigned char rowL[336];
    __shared__ unsigned char colL[336];

    const int b   = blockIdx.x;
    const int tid = threadIdx.x;

    // ---- small param staging (padded to 16 so N-padding lanes contribute 0) ----
    if (tid < 16) {
        biasL[tid] = (tid < 12) ? attn_b[tid] : 0.0f;
        projL[tid] = (tid < 12) ? proj_W[tid] : 0.0f;
    }
    if (tid < 64) embL[N_SPARSE * EMB_DIM + tid] = 0.0f;   // zero pad row

    // ---- pair index tables (triu_indices order), padded with (26,26) ----
    for (int p = tid; p < 336; p += 256) {
        int r = N_SPARSE, c = N_SPARSE;   // pad -> zero row
        if (p < N_PAIR) {
            int off = 0;
            for (int i = 0; i < N_SPARSE - 1; ++i) {
                int cnt = N_SPARSE - 1 - i;
                if (p < off + cnt) { r = i; c = i + 1 + (p - off); break; }
                off += cnt;
            }
        }
        rowL[p] = (unsigned char)r;
        colL[p] = (unsigned char)c;
    }

    // ---- async gather: 26 embedding rows (416 x 16B) + attn_W (192 x 16B) -> LDS ----
    for (int u = tid; u < 608; u += 256) {
        if (u < 416) {
            int f = u >> 4;
            int k = u & 15;
            int id = sparse_idx[b * N_SPARSE + f];
            const float* src = emb_tables + ((size_t)f * VOCAB + (size_t)id) * EMB_DIM + k * 4;
            async_copy16((uint32_t)(uintptr_t)&embL[f * EMB_DIM + k * 4], src);
        } else {
            int u2 = u - 416;
            async_copy16((uint32_t)(uintptr_t)&WsL[u2 * 4], attn_W + u2 * 4);
        }
    }
    asm volatile("s_wait_asynccnt 0" ::: "memory");
    __syncthreads();

    // ---- pass A: attention GEMM [325x64] x [64x12] via v_wmma_f32_16x16x32_f16 ----
    const int lane  = tid & 31;
    const int wave  = tid >> 5;
    const int grp   = lane >> 4;   // half-wave group
    const int laneN = lane & 15;   // N column in B/C layout
    const float pb  = proj_b[0];

    // hoisted per-lane epilogue constants
    const float bias_n = biasL[laneN];
    const float proj_n = projL[laneN];

    // B operands (32x16 f16 layout): half i of VGPR block -> K = i + 8*((i>=8)+grp) ; col N = laneN
    v16h b0, b1;
    #pragma unroll
    for (int i = 0; i < 16; ++i) {
        int k0 = i + 8 * ((i >= 8) + grp);
        float w0 = (laneN < 12) ? WsL[k0 * 12 + laneN] : 0.0f;
        float w1 = (laneN < 12) ? WsL[(32 + k0) * 12 + laneN] : 0.0f;
        b0[i] = (_Float16)w0;
        b1[i] = (_Float16)w1;
    }

    #pragma unroll
    for (int it = 0; it < 3; ++it) {
        int t = wave + it * 8;
        t = (t > N_TILE - 1) ? (N_TILE - 1) : t;   // uniform clamp; dup writes benign
        const int p0 = t * 16;
        const int p  = p0 + laneN;                 // A-matrix row M = laneN
        const int r  = rowL[p];
        const int c  = colL[p];

        // A (16x32 f16): K = i + 8*((i>=8)+grp) -> contiguous 8-float runs, b128 loads
        const float* er = &embL[r * EMB_DIM + 8 * grp];
        const float* ec = &embL[c * EMB_DIM + 8 * grp];
        f4 r0 = *(const f4*)(er +  0), c0 = *(const f4*)(ec +  0);  // K=8g+0..3
        f4 r1 = *(const f4*)(er +  4), c1 = *(const f4*)(ec +  4);  // K=8g+4..7
        f4 r2 = *(const f4*)(er + 16), c2 = *(const f4*)(ec + 16);  // K=16+8g..
        f4 r3 = *(const f4*)(er + 20), c3 = *(const f4*)(ec + 20);
        f4 r4 = *(const f4*)(er + 32), c4 = *(const f4*)(ec + 32);  // chunk 1
        f4 r5 = *(const f4*)(er + 36), c5 = *(const f4*)(ec + 36);
        f4 r6 = *(const f4*)(er + 48), c6 = *(const f4*)(ec + 48);
        f4 r7 = *(const f4*)(er + 52), c7 = *(const f4*)(ec + 52);

        v16h a0, a1;
        #pragma unroll
        for (int j = 0; j < 4; ++j) {
            a0[j]      = (_Float16)(r0[j] * c0[j]);
            a0[4 + j]  = (_Float16)(r1[j] * c1[j]);
            a0[8 + j]  = (_Float16)(r2[j] * c2[j]);
            a0[12 + j] = (_Float16)(r3[j] * c3[j]);
            a1[j]      = (_Float16)(r4[j] * c4[j]);
            a1[4 + j]  = (_Float16)(r5[j] * c5[j]);
            a1[8 + j]  = (_Float16)(r6[j] * c6[j]);
            a1[12 + j] = (_Float16)(r7[j] * c7[j]);
        }

        v8f acc = {};
        acc = __builtin_amdgcn_wmma_f32_16x16x32_f16(false, a0, false, b0,
                                                     (short)0, acc, false, false);
        acc = __builtin_amdgcn_wmma_f32_16x16x32_f16(false, a1, false, b1,
                                                     (short)0, acc, false, false);

        // epilogue: bias + ReLU + dot with proj_W over N (16 lanes of the group)
        #pragma unroll
        for (int j = 0; j < 8; ++j) {
            float v = fmaxf(acc[j] + bias_n, 0.0f) * proj_n;
            v = group16_sum(v);
            if (laneN == 0)   // lanes 0 & 16: C/D VGPR j holds M = grp*8 + j
                logitsL[p0 + grp * 8 + j] = (v + pb) * TEMP_INV;
        }
    }
    __syncthreads();

    // ---- softmax over 325 pairs ----
    float lm = -1e30f;
    for (int p = tid; p < N_PAIR; p += 256) lm = fmaxf(lm, logitsL[p]);
    red[tid] = lm;
    __syncthreads();
    for (int s = 128; s > 0; s >>= 1) {
        if (tid < s) red[tid] = fmaxf(red[tid], red[tid + s]);
        __syncthreads();
    }
    const float mx = red[0];
    __syncthreads();

    float ls = 0.0f;
    for (int p = tid; p < N_PAIR; p += 256) {
        float e = __expf(logitsL[p] - mx);
        expL[p] = e;
        ls += e;
    }
    red[tid] = ls;
    __syncthreads();
    for (int s = 128; s > 0; s >>= 1) {
        if (tid < s) red[tid] += red[tid + s];
        __syncthreads();
    }
    const float invSum = 1.0f / red[0];

    // ---- pass B: att_out[d] = sum_p score[p] * emb[r,d]*emb[c,d] (recompute cross) ----
    const int d = tid & 63;
    const int chunk = tid >> 6;
    float att = 0.0f;
    for (int p = chunk; p < N_PAIR; p += 4) {
        float w = expL[p];
        int r = rowL[p], c = colL[p];
        att += w * embL[r * EMB_DIM + d] * embL[c * EMB_DIM + d];
    }
    attRed[chunk * 64 + d] = att;
    __syncthreads();

    if (tid < 64) {
        float s = attRed[tid] + attRed[64 + tid] + attRed[128 + tid] + attRed[192 + tid];
        s2[tid] = s * invSum * pred_W[tid];
    }
    __syncthreads();

    if (tid == 0) {
        float p2 = pred_b[0];
        #pragma unroll
        for (int k = 0; k < 64; ++k) p2 += s2[k];
        float p1 = lin_b[0];
        for (int k = 0; k < N_DENSE; ++k)
            p1 += dense_x[b * N_DENSE + k] * lin_W[k];
        for (int k = 0; k < N_SPARSE; ++k)
            p1 += (float)sparse_idx[b * N_SPARSE + k] * lin_W[N_DENSE + k];
        float z = p1 + p2;
        out[b] = 1.0f / (1.0f + __expf(-z));
    }
}

extern "C" void kernel_launch(void* const* d_in, const int* in_sizes, int n_in,
                              void* d_out, int out_size, void* d_ws, size_t ws_size,
                              hipStream_t stream) {
    (void)n_in; (void)d_ws; (void)ws_size; (void)out_size;
    const float* dense_x    = (const float*)d_in[0];
    const int*   sparse_idx = (const int*)  d_in[1];
    const float* emb_tables = (const float*)d_in[2];
    const float* attn_W     = (const float*)d_in[3];
    const float* attn_b     = (const float*)d_in[4];
    const float* proj_W     = (const float*)d_in[5];
    const float* proj_b     = (const float*)d_in[6];
    const float* lin_W      = (const float*)d_in[7];
    const float* lin_b      = (const float*)d_in[8];
    const float* pred_W     = (const float*)d_in[9];
    const float* pred_b     = (const float*)d_in[10];
    float* out = (float*)d_out;

    const int batch = in_sizes[0] / N_DENSE;   // 8192
    afm_fused_kernel<<<batch, 256, 0, stream>>>(
        dense_x, sparse_idx, emb_tables, attn_W, attn_b, proj_W, proj_b,
        lin_W, lin_b, pred_W, pred_b, out);
}
